// MultiHeadSelfAttention_5471788335242
// MI455X (gfx1250) — matrix operand order repaired and checked by reading, not compiled, not run on hardware
//
#include <hip/hip_runtime.h>
#include <hip/hip_bf16.h>

#define D_MODEL 1024
#define N_HEADS 16
#define D_HEAD  64
#define BATCH   4
#define SEQ     2048
#define M_ROWS  (BATCH * SEQ)   // 8192

typedef __attribute__((ext_vector_type(16))) _Float16 v16h;
typedef __attribute__((ext_vector_type(8)))  _Float16 v8h;
typedef __attribute__((ext_vector_type(4)))  _Float16 v4h;
typedef __attribute__((ext_vector_type(8)))  float    v8f;
typedef int v4i __attribute__((vector_size(16)));   // matches builtin param type

#define GLOBAL_AS __attribute__((address_space(1)))
#define LDS_AS    __attribute__((address_space(3)))

// ---- async global->LDS copy (ASYNCcnt path), with safe fallback ----------
#if defined(__gfx1250__) && __has_builtin(__builtin_amdgcn_global_load_async_to_lds_b128)
#define HAVE_ASYNC_LDS 1
__device__ __forceinline__ void async_copy16(const _Float16* g, _Float16* l) {
    __builtin_amdgcn_global_load_async_to_lds_b128(
        (GLOBAL_AS v4i*)g, (LDS_AS v4i*)l, /*offset=*/0, /*cpol=*/0);
}
__device__ __forceinline__ void wait_async0() {
#if __has_builtin(__builtin_amdgcn_s_wait_asynccnt)
    __builtin_amdgcn_s_wait_asynccnt(0);
#else
    asm volatile("s_wait_asynccnt 0" ::: "memory");
#endif
}
#endif

__device__ __forceinline__ v16h cat16(v8h lo, v8h hi) {
    return __builtin_shufflevector(lo, hi, 0,1,2,3,4,5,6,7,8,9,10,11,12,13,14,15);
}

__device__ __forceinline__ v8f wmma_f16(v16h a, v16h b, v8f c) {
    // D = A(16x32 f16) * B(32x16 f16) + C(16x16 f32)
    return __builtin_amdgcn_wmma_f32_16x16x32_f16(
        /*neg_a=*/false, a, /*neg_b=*/false, b,
        /*c_mod=*/(short)0, c, /*reuse_a=*/false, /*reuse_b=*/false);
}

// ---------------------------------------------------------------------------
// fp32 -> fp16 conversion (bandwidth-trivial: ~56MB total)
// ---------------------------------------------------------------------------
__global__ __launch_bounds__(256) void cvt_f32_f16(const float* __restrict__ in,
                                                   _Float16* __restrict__ out,
                                                   int n) {
    int i = (blockIdx.x * 256 + threadIdx.x) * 4;
    if (i + 3 < n) {
        float4 v = *(const float4*)(in + i);
        v4h h;
        h[0] = (_Float16)v.x; h[1] = (_Float16)v.y;
        h[2] = (_Float16)v.z; h[3] = (_Float16)v.w;
        *(v4h*)(out + i) = h;
    }
}

// ---------------------------------------------------------------------------
// GEMM: C[M_ROWS, D_MODEL] = A[M_ROWS, D_MODEL] * W^T, W row-major [N,K].
// Block tile 128x128, K-step 32, double-buffered LDS, 8 waves in 4(M)x2(N),
// each wave computes 32x64 = 2x4 WMMA tiles. One barrier per K-step; the
// next K-slice is staged (async when available) while WMMAs run.
// ---------------------------------------------------------------------------
#define BM  128
#define BN  128
#define BK  32
#define LDA 40   // BK + 8 f16 pad (80B row stride, 16B aligned)

// Each thread stages one 32B row-chunk of A and of B per K-step.
__device__ __forceinline__ void stage_tile(const _Float16* ga, const _Float16* gb,
                                           _Float16* la, _Float16* lb) {
#if defined(HAVE_ASYNC_LDS)
    async_copy16(ga,     la);
    async_copy16(ga + 8, la + 8);
    async_copy16(gb,     lb);
    async_copy16(gb + 8, lb + 8);
#else
    v8h a0 = *(const v8h*)ga;
    v8h a1 = *(const v8h*)(ga + 8);
    v8h b0 = *(const v8h*)gb;
    v8h b1 = *(const v8h*)(gb + 8);
    *(v8h*)la       = a0;
    *(v8h*)(la + 8) = a1;
    *(v8h*)lb       = b0;
    *(v8h*)(lb + 8) = b1;
#endif
}

template<int OUT_F16>
__global__ __launch_bounds__(256) void gemm_nt(const _Float16* __restrict__ A,
                                               const _Float16* __restrict__ W,
                                               _Float16* __restrict__ outh,
                                               float*    __restrict__ outf) {
    __shared__ __align__(16) _Float16 Ash[2][BM * LDA];
    __shared__ __align__(16) _Float16 Bsh[2][BN * LDA];

    const int tid    = threadIdx.x;
    const int wave   = tid >> 5;
    const int lane   = tid & 31;
    const int laneLo = lane & 15;
    const int laneHi = lane >> 4;
    const int wm     = wave >> 1;       // 0..3
    const int wn     = wave & 1;        // 0..1
    const int m0     = blockIdx.y * BM;
    const int n0     = blockIdx.x * BN;

    const int ldr  = tid >> 1;          // 0..127: row staged by this thread
    const int ldc  = (tid & 1) * 16;    // 0 or 16: K sub-chunk
    const int lofs = ldr * LDA + ldc;

    const _Float16* ga0 = A + (size_t)(m0 + ldr) * D_MODEL + ldc;
    const _Float16* gb0 = W + (size_t)(n0 + ldr) * D_MODEL + ldc;

    v8f zf = {};
    v8f acc[2][4];
    #pragma unroll
    for (int i = 0; i < 2; ++i)
        #pragma unroll
        for (int j = 0; j < 4; ++j) acc[i][j] = zf;

    // prologue: stage K-slice 0 into buffer 0
    stage_tile(ga0, gb0, &Ash[0][lofs], &Bsh[0][lofs]);

    const int NK = D_MODEL / BK;        // 32
    for (int i = 0; i < NK; ++i) {
        const int buf = i & 1;
#if defined(HAVE_ASYNC_LDS)
        wait_async0();                  // my staged data landed in LDS
#endif
        __syncthreads();                // all waves' staging visible; prev reads done

        if (i + 1 < NK) {
            // stage next K-slice into the other buffer while we compute
            stage_tile(ga0 + (i + 1) * BK, gb0 + (i + 1) * BK,
                       &Ash[buf ^ 1][lofs], &Bsh[buf ^ 1][lofs]);
#if !defined(HAVE_ASYNC_LDS)
            __builtin_prefetch(ga0 + (i + 2) * BK, 0, 1);
            __builtin_prefetch(gb0 + (i + 2) * BK, 0, 1);
#endif
        }

        // ---- fragments from current buffer ----
        const _Float16* As = Ash[buf];
        const _Float16* Bs = Bsh[buf];
        v16h af[2], bf[4];
        #pragma unroll
        for (int mt = 0; mt < 2; ++mt) {
            // A layout: lane(lo,hi) holds row=lo, K = hi*8+0..7 and hi*8+16..23
            const _Float16* p = &As[(wm * 32 + mt * 16 + laneLo) * LDA + laneHi * 8];
            af[mt] = cat16(*(const v8h*)p, *(const v8h*)(p + 16));
        }
        #pragma unroll
        for (int nt = 0; nt < 4; ++nt) {
            // B layout: lane(lo,hi) holds col=lo, K = hi*8+0..7 and hi*8+16..23
            const _Float16* p = &Bs[(wn * 64 + nt * 16 + laneLo) * LDA + laneHi * 8];
            bf[nt] = cat16(*(const v8h*)p, *(const v8h*)(p + 16));
        }

        #pragma unroll
        for (int mt = 0; mt < 2; ++mt)
            #pragma unroll
            for (int nt = 0; nt < 4; ++nt)
                acc[mt][nt] = wmma_f16(af[mt], bf[nt], acc[mt][nt]);
    }

    // ---- epilogue: C layout row = r + laneHi*8, col = laneLo ----
    #pragma unroll
    for (int mt = 0; mt < 2; ++mt)
        #pragma unroll
        for (int nt = 0; nt < 4; ++nt)
            #pragma unroll
            for (int r = 0; r < 8; ++r) {
                int row = m0 + wm * 32 + mt * 16 + r + laneHi * 8;
                int col = n0 + wn * 64 + nt * 16 + laneLo;
                if (OUT_F16) outh[(size_t)row * D_MODEL + col] = (_Float16)acc[mt][nt][r];
                else         outf[(size_t)row * D_MODEL + col] = acc[mt][nt][r];
            }
}

// ---------------------------------------------------------------------------
// Flash attention, causal. One block = 128 queries of one (b,h);
// 8 waves x 16 queries. Stream 64-key blocks; online softmax in f32.
// ---------------------------------------------------------------------------
#define AT_BQ  128
#define AT_BKV 64
#define VTS    72   // V^T LDS row stride (keys dim, padded)
#define PS     72   // P LDS row stride

__global__ __launch_bounds__(256) void attn_fwd(const _Float16* __restrict__ Qh,
                                                const _Float16* __restrict__ Kh,
                                                const _Float16* __restrict__ Vh,
                                                _Float16* __restrict__ Oh) {
    __shared__ __align__(16) _Float16 VT[D_HEAD * VTS];        // V^T: [64 d][72 keys]
    __shared__ __align__(16) _Float16 Psh[8 * 16 * PS];        // per-wave P: [16 q][72 k]

    const int tid    = threadIdx.x;
    const int wave   = tid >> 5;
    const int lane   = tid & 31;
    const int laneLo = lane & 15;
    const int laneHi = lane >> 4;
    const int b  = blockIdx.z;
    const int h  = blockIdx.y;
    const int q0 = blockIdx.x * AT_BQ;

    const _Float16* Qb = Qh + (size_t)b * SEQ * D_MODEL + h * D_HEAD;
    const _Float16* Kb = Kh + (size_t)b * SEQ * D_MODEL + h * D_HEAD;
    const _Float16* Vb = Vh + (size_t)b * SEQ * D_MODEL + h * D_HEAD;

    // Q A-fragments (kept in registers for the whole kernel): K-dim = d = 64
    v16h aQ[2];
    {
        int q = q0 + wave * 16 + laneLo;
        #pragma unroll
        for (int ks = 0; ks < 2; ++ks) {
            const _Float16* p = Qb + (size_t)q * D_MODEL + ks * 32 + laneHi * 8;
            aQ[ks] = cat16(*(const v8h*)p, *(const v8h*)(p + 16));
        }
    }

    v8f zf = {};
    v8f O[4]; // output accum: 16 q x 64 d
    #pragma unroll
    for (int nt = 0; nt < 4; ++nt) O[nt] = zf;
    float mrow[8], lrow[8];
    #pragma unroll
    for (int r = 0; r < 8; ++r) { mrow[r] = -1e30f; lrow[r] = 0.f; }
    const float scale = 0.125f;  // 1/sqrt(64)

    const int kend = q0 + AT_BQ; // causal: keys in [0, q0+127]
    for (int kb = 0; kb < kend; kb += AT_BKV) {
        __syncthreads();
        // ---- stage V^T into LDS (transpose on store) ----
        {
            int key = tid >> 2;            // 0..63
            int dc  = (tid & 3) * 16;      // 0,16,32,48
            const _Float16* vp = Vb + (size_t)(kb + key) * D_MODEL + dc;
            v8h v0 = *(const v8h*)vp;
            v8h v1 = *(const v8h*)(vp + 8);
            #pragma unroll
            for (int e = 0; e < 8; ++e) {
                VT[(dc + e)     * VTS + key] = v0[e];
                VT[(dc + 8 + e) * VTS + key] = v1[e];
            }
            if (kb + AT_BKV < kend) {      // hint next K/V block toward L2/L0
                __builtin_prefetch(vp + (size_t)AT_BKV * D_MODEL, 0, 1);
                __builtin_prefetch(Kb + (size_t)(kb + AT_BKV + key) * D_MODEL + dc, 0, 1);
            }
        }
        __syncthreads();

        // ---- S = (Q K^T) for 16q x 64k ----
        v8f S[4];
        #pragma unroll
        for (int nt = 0; nt < 4; ++nt) S[nt] = zf;
        #pragma unroll
        for (int ks = 0; ks < 2; ++ks) {
            v16h bK[4];
            #pragma unroll
            for (int nt = 0; nt < 4; ++nt) {
                // B frag straight from global: K rows are contiguous d values
                const _Float16* p = Kb + (size_t)(kb + nt * 16 + laneLo) * D_MODEL
                                       + ks * 32 + laneHi * 8;
                bK[nt] = cat16(*(const v8h*)p, *(const v8h*)(p + 16));
            }
            #pragma unroll
            for (int nt = 0; nt < 4; ++nt) S[nt] = wmma_f16(aQ[ks], bK[nt], S[nt]);
        }

        // ---- scale + causal mask (C layout: row = r + laneHi*8, col = laneLo) ----
        #pragma unroll
        for (int nt = 0; nt < 4; ++nt) {
            int col = kb + nt * 16 + laneLo;
            #pragma unroll
            for (int r = 0; r < 8; ++r) {
                int qi = q0 + wave * 16 + r + laneHi * 8;
                float s = S[nt][r] * scale;
                S[nt][r] = (col > qi) ? -1e30f : s;
            }
        }

        // ---- online softmax: row reductions across 16 lanes of each half ----
        #pragma unroll
        for (int r = 0; r < 8; ++r) {
            float t = fmaxf(fmaxf(S[0][r], S[1][r]), fmaxf(S[2][r], S[3][r]));
            t = fmaxf(t, __shfl_xor(t, 1, 16));
            t = fmaxf(t, __shfl_xor(t, 2, 16));
            t = fmaxf(t, __shfl_xor(t, 4, 16));
            t = fmaxf(t, __shfl_xor(t, 8, 16));
            float mn = fmaxf(mrow[r], t);
            float f  = __expf(mrow[r] - mn);
            lrow[r] *= f;
            #pragma unroll
            for (int nt = 0; nt < 4; ++nt) O[nt][r] *= f;
            mrow[r] = mn;
        }

        // ---- P = exp(S - m); write per-wave P tile to LDS; accumulate l ----
        _Float16* prow = &Psh[wave * 16 * PS];
        #pragma unroll
        for (int r = 0; r < 8; ++r) {
            float rs = 0.f;
            #pragma unroll
            for (int nt = 0; nt < 4; ++nt) {
                float p = __expf(S[nt][r] - mrow[r]);
                rs += p;
                prow[(r + laneHi * 8) * PS + nt * 16 + laneLo] = (_Float16)p;
            }
            rs += __shfl_xor(rs, 1, 16);
            rs += __shfl_xor(rs, 2, 16);
            rs += __shfl_xor(rs, 4, 16);
            rs += __shfl_xor(rs, 8, 16);
            lrow[r] += rs;
        }
        // per-wave LDS RAW: make P writes visible before A-layout reads
        asm volatile("s_wait_dscnt 0" ::: "memory");

        // ---- O += P V (A from Psh in A-layout, B from V^T in LDS) ----
        #pragma unroll
        for (int ks = 0; ks < 2; ++ks) {
            const _Float16* pp = prow + laneLo * PS + ks * 32 + laneHi * 8;
            v16h aP = cat16(*(const v8h*)pp, *(const v8h*)(pp + 16));
            #pragma unroll
            for (int nt = 0; nt < 4; ++nt) {
                const _Float16* vp = &VT[(nt * 16 + laneLo) * VTS + ks * 32 + laneHi * 8];
                v16h bV = cat16(*(const v8h*)vp, *(const v8h*)(vp + 16));
                O[nt] = wmma_f16(aP, bV, O[nt]);
            }
        }
    }

    // ---- finalize: O / l, store f16 attention output (heads -> columns) ----
    _Float16* ob = Oh + (size_t)(b * SEQ + q0 + wave * 16) * D_MODEL + h * D_HEAD;
    #pragma unroll
    for (int nt = 0; nt < 4; ++nt)
        #pragma unroll
        for (int r = 0; r < 8; ++r) {
            float v = O[nt][r] / lrow[r];
            ob[(size_t)(r + laneHi * 8) * D_MODEL + nt * 16 + laneLo] = (_Float16)v;
        }
}

// ---------------------------------------------------------------------------
extern "C" void kernel_launch(void* const* d_in, const int* in_sizes, int n_in,
                              void* d_out, int out_size, void* d_ws, size_t ws_size,
                              hipStream_t stream) {
    (void)in_sizes; (void)n_in; (void)out_size; (void)ws_size;
    const float* x  = (const float*)d_in[0];
    const float* Wq = (const float*)d_in[1];
    const float* Wk = (const float*)d_in[2];
    const float* Wv = (const float*)d_in[3];
    const float* Wo = (const float*)d_in[4];
    float* out = (float*)d_out;

    char* ws = (char*)d_ws;
    size_t off = 0;
    auto alloc = [&](size_t bytes) -> void* {
        void* p = ws + off;
        off += (bytes + 255) & ~(size_t)255;
        return p;
    };
    const size_t xhB = (size_t)M_ROWS * D_MODEL * sizeof(_Float16);
    const size_t wB  = (size_t)D_MODEL * D_MODEL * sizeof(_Float16);
    _Float16* xh  = (_Float16*)alloc(xhB);
    _Float16* Wqh = (_Float16*)alloc(wB);
    _Float16* Wkh = (_Float16*)alloc(wB);
    _Float16* Wvh = (_Float16*)alloc(wB);
    _Float16* Woh = (_Float16*)alloc(wB);
    _Float16* Qh  = (_Float16*)alloc(xhB);
    _Float16* Kh  = (_Float16*)alloc(xhB);
    _Float16* Vh  = (_Float16*)alloc(xhB);
    _Float16* Ah  = (_Float16*)alloc(xhB);

    // fp32 -> fp16
    int nx = M_ROWS * D_MODEL;         // 8,388,608
    int nw = D_MODEL * D_MODEL;        // 1,048,576
    cvt_f32_f16<<<nx / 1024, 256, 0, stream>>>(x,  xh,  nx);
    cvt_f32_f16<<<nw / 1024, 256, 0, stream>>>(Wq, Wqh, nw);
    cvt_f32_f16<<<nw / 1024, 256, 0, stream>>>(Wk, Wkh, nw);
    cvt_f32_f16<<<nw / 1024, 256, 0, stream>>>(Wv, Wvh, nw);
    cvt_f32_f16<<<nw / 1024, 256, 0, stream>>>(Wo, Woh, nw);

    // Q/K/V projections
    dim3 gg(D_MODEL / BN, M_ROWS / BM);   // (8, 64)
    gemm_nt<1><<<gg, 256, 0, stream>>>(xh, Wqh, Qh, nullptr);
    gemm_nt<1><<<gg, 256, 0, stream>>>(xh, Wkh, Kh, nullptr);
    gemm_nt<1><<<gg, 256, 0, stream>>>(xh, Wvh, Vh, nullptr);

    // causal flash attention
    dim3 ga(SEQ / AT_BQ, N_HEADS, BATCH); // (16, 16, 4)
    attn_fwd<<<ga, 256, 0, stream>>>(Qh, Kh, Vh, Ah);

    // output projection -> fp32
    gemm_nt<0><<<gg, 256, 0, stream>>>(Ah, Woh, nullptr, out);
}
